// SelfAttention_85487029059932
// MI455X (gfx1250) — matrix operand order repaired and checked by reading, not compiled, not run on hardware
//
#include <hip/hip_runtime.h>

typedef __attribute__((ext_vector_type(16))) __bf16 v16bf;
typedef __attribute__((ext_vector_type(8)))  __bf16 v8bf;
typedef __attribute__((ext_vector_type(2)))  __bf16 v2bf;
typedef __attribute__((ext_vector_type(8)))  float  v8f;

#define BB    16
#define CC    64
#define NN    4096
#define MIDD  8
#define WAVES 8   // waves per block in attention kernel

// softmax scale folded into q at projection time: (1/sqrt(MID)) * log2(e)
#define QSCALE 0.51006364f

__device__ __forceinline__ v16bf zero16() {
  v16bf v;
  #pragma unroll
  for (int i = 0; i < 16; ++i) v[i] = (__bf16)0.0f;
  return v;
}

// P-buffer row offset (bf16 elements).  Rows are 48 elems (96B, 32B-aligned);
// rows 8..15 live +64B further so lane i / lane i+16 stores use disjoint banks.
__device__ __forceinline__ int prow_off(int row) {
  return (row & 7) * 48 + (row >> 3) * 416;
}
#define PBUF_ELEMS 800   // per-wave P buffer size (rows 8..15 end at 784)
#define OROW 65          // padded O row (f32) to break bank conflicts

// ---------------------------------------------------------------------------
// Kernel 1: 1x1-conv projections (bf16 staging for WMMA).
//   qT  : [B][N][8]    q pre-scaled by QSCALE, row-contiguous 16B A-rows
//   kT16: [B][N][16]   k padded with zeros to 32B rows (direct B-tile loads)
//   vB  : [B][C][N]    key positions swizzled within each 32-block to match
//                      the packed-P K-slot permutation (see attn kernel)
//   zbuf: 128B zeros (B-tile source for lanes carrying K=16..31)
// x is read once into VGPRs; W streams via wave-uniform (scalar) loads.
// 256 threads: lower 128 do q,k + v[0..31]; upper 128 do v[32..63] for the
// same 128 positions (doubles wave parallelism for latency hiding).
// ---------------------------------------------------------------------------
__global__ __launch_bounds__(256) void proj_kernel(
    const float* __restrict__ x,  const float* __restrict__ Wq,
    const float* __restrict__ Wk, const float* __restrict__ Wv,
    __bf16* __restrict__ qT, __bf16* __restrict__ kT16,
    __bf16* __restrict__ vB, __bf16* __restrict__ zbuf)
{
  const int t   = threadIdx.x;
  const int tid = t & 127;
  const int hi  = t >> 7;                       // 0: q,k,v[0..31]  1: v[32..63]
  if (blockIdx.x == 0 && t < 64) zbuf[t] = (__bf16)0.0f;

  const int b = blockIdx.x / (NN / 128);
  const int n = (blockIdx.x % (NN / 128)) * 128 + tid;

  float xr[CC];
  const float* xp = x + (size_t)b * CC * NN + n;
  #pragma unroll
  for (int c = 0; c < CC; ++c) xr[c] = xp[(size_t)c * NN];   // coalesced

  if (!hi) {
    __bf16* qp = qT   + ((size_t)b * NN + n) * MIDD;
    __bf16* kp = kT16 + ((size_t)b * NN + n) * 16;
    #pragma unroll 1
    for (int m = 0; m < MIDD; ++m) {
      float qa = 0.0f, ka = 0.0f;
      #pragma unroll
      for (int c = 0; c < CC; ++c) {
        qa += Wq[m * CC + c] * xr[c];     // uniform address -> s_load
        ka += Wk[m * CC + c] * xr[c];
      }
      qp[m]        = (__bf16)(qa * QSCALE);
      kp[m]        = (__bf16)ka;
      kp[m + MIDD] = (__bf16)0.0f;        // zero pad K=8..15
    }
  }

  // key swizzle within each 32-block: logical key k -> phys slot
  //   jj=k&15, par=k>>4 : phys = 2*jj + par + ((jj&4)<<1) - (jj&8)
  const int jj   = n & 15;
  const int par  = (n >> 4) & 1;
  const int phys = 2 * jj + par + ((jj & 4) << 1) - (jj & 8);
  const int nswz = (n & ~31) | phys;

  __bf16* vp = vB + (size_t)b * CC * NN + nswz;
  const int obase = hi * 32;
  #pragma unroll 2
  for (int oo = 0; oo < 32; ++oo) {
    const int o = obase + oo;
    float va = 0.0f;
    #pragma unroll
    for (int c = 0; c < CC; ++c) va += Wv[o * CC + c] * xr[c];
    vp[(size_t)o * NN] = (__bf16)va;
  }
}

// ---------------------------------------------------------------------------
// Kernel 2: fused flash-attention without online-max (logits provably tiny;
// softmax is shift-invariant and exp2 stays far from fp32 overflow).
//   S = (Q*QSCALE) Kt (log2 domain) ; P = exp2(S) ; O += P V
//   denominator accumulated by a 5th WMMA against a ones-column B tile.
// P transpose through LDS uses a permuted K order so each lane's (p0,p1)
// pair packs into one v_cvt_pk_bf16_f32 + ds_store_b32, and the A operand
// reloads as a single 32B ds_load.  V was pre-swizzled to the same K order.
// ---------------------------------------------------------------------------
__global__ __launch_bounds__(WAVES * 32) void attn_kernel(
    const float* __restrict__ x, const float* __restrict__ gamma,
    const __bf16* __restrict__ qT, const __bf16* __restrict__ kT16,
    const __bf16* __restrict__ vB, const __bf16* __restrict__ zbuf,
    float* __restrict__ out)
{
  __shared__ __align__(32) __bf16 p_s[WAVES][PBUF_ELEMS];
  __shared__ __align__(16) float  o_s[WAVES][16 * OROW];

  const int lane = threadIdx.x & 31;
  const int wave = threadIdx.x >> 5;
  const int tile = blockIdx.x * WAVES + wave;   // global 16-row query tile
  const int b    = tile >> 8;                   // 256 tiles per batch
  const int m0   = (tile & 255) * 16;
  const int half = lane >> 4;                   // 0: lanes 0-15, 1: lanes 16-31
  const int l15  = lane & 15;

  // --- Q as 16x32 bf16 A-matrix (real K=0..7; rest zero-padded) ----------
  v16bf aq = zero16();
  if (!half) {
    v8bf qv = *(const v8bf*)(qT + ((size_t)b * NN + m0 + l15) * MIDD);
    #pragma unroll
    for (int e = 0; e < 8; ++e) aq[e] = qv[e];
  }

  // --- ones-column B tile (permutation-invariant): col 0 all ones --------
  v16bf bones = zero16();
  if (l15 == 0) {
    #pragma unroll
    for (int e = 0; e < 16; ++e) bones[e] = (__bf16)1.0f;
  }

  v8f O0 = {}, O1 = {}, O2 = {}, O3 = {}, O4 = {};   // O4 col0 = row sums

  // K B-tile pointers: lanes 0-15 walk padded kT16 rows; lanes 16-31 stay on
  // a zero buffer (they carry the K=16..31 zero padding).  One 32B load each.
  const __bf16* kp0 = half ? zbuf : (kT16 + ((size_t)b * NN + l15) * 16);
  const __bf16* kp1 = half ? zbuf : (kT16 + ((size_t)b * NN + 16 + l15) * 16);
  const size_t  kstep = half ? 0 : (size_t)32 * 16;   // elems per 32-key iter

  const __bf16* vbase = vB + (size_t)b * CC * NN;
  const int koff = half * 16;                  // V B-tiles: lanes>=16 = K 16..31
  const int pst  = l15 * 2 + ((l15 & 4) << 1) - (l15 & 8);  // packed (p0,p1) column slot
  const int pld  = prow_off(l15) + half * 16;  // A reload: 16 contiguous elems

  for (int n0 = 0; n0 < NN; n0 += 32) {
    v16bf bk0 = *(const v16bf*)kp0;
    v16bf bk1 = *(const v16bf*)kp1;
    kp0 += kstep; kp1 += kstep;

    v8f s0 = {}, s1 = {};
    s0 = __builtin_amdgcn_wmma_f32_16x16x32_bf16(false, aq, false, bk0,
                                                 (short)0, s0, false, false);
    s1 = __builtin_amdgcn_wmma_f32_16x16x32_bf16(false, aq, false, bk1,
                                                 (short)0, s1, false, false);

    // --- P = exp2(S): packed pair store, K-permuted transpose ------------
    #pragma unroll
    for (int r = 0; r < 8; ++r) {
      const int row = r + 8 * half;             // D layout: comp r -> row r(+8)
      v2bf pk;
      pk[0] = (__bf16)__builtin_amdgcn_exp2f(s0[r]);   // key n0+l15
      pk[1] = (__bf16)__builtin_amdgcn_exp2f(s1[r]);   // key n0+16+l15
      *(v2bf*)(&p_s[wave][prow_off(row) + pst]) = pk;
    }
    asm volatile("s_wait_dscnt 0x0" ::: "memory");  // same-wave LDS RAW

    v16bf ap = *(const v16bf*)(&p_s[wave][pld]);     // one 32B A-operand load

    // --- O += P * Vt : 4 WMMAs for 64 channels + 1 for the denominator ---
    {
      v16bf bv = *(const v16bf*)(vbase + (size_t)(0 * 16 + l15) * NN + n0 + koff);
      O0 = __builtin_amdgcn_wmma_f32_16x16x32_bf16(false, ap, false, bv,
                                                   (short)0, O0, false, false);
    }
    {
      v16bf bv = *(const v16bf*)(vbase + (size_t)(1 * 16 + l15) * NN + n0 + koff);
      O1 = __builtin_amdgcn_wmma_f32_16x16x32_bf16(false, ap, false, bv,
                                                   (short)0, O1, false, false);
    }
    {
      v16bf bv = *(const v16bf*)(vbase + (size_t)(2 * 16 + l15) * NN + n0 + koff);
      O2 = __builtin_amdgcn_wmma_f32_16x16x32_bf16(false, ap, false, bv,
                                                   (short)0, O2, false, false);
    }
    {
      v16bf bv = *(const v16bf*)(vbase + (size_t)(3 * 16 + l15) * NN + n0 + koff);
      O3 = __builtin_amdgcn_wmma_f32_16x16x32_bf16(false, ap, false, bv,
                                                   (short)0, O3, false, false);
    }
    O4 = __builtin_amdgcn_wmma_f32_16x16x32_bf16(false, ap, false, bones,
                                                 (short)0, O4, false, false);
  }

  // --- epilogue: normalize by l (col 0 of O4), stage in LDS, write back --
  #pragma unroll
  for (int r = 0; r < 8; ++r) {
    const float lr  = __shfl(O4[r], half * 16, 32);  // broadcast col-0 lane
    const float inv = __builtin_amdgcn_rcpf(lr);
    const int row = r + 8 * half;
    o_s[wave][row * OROW + 0 * 16 + l15] = O0[r] * inv;
    o_s[wave][row * OROW + 1 * 16 + l15] = O1[r] * inv;
    o_s[wave][row * OROW + 2 * 16 + l15] = O2[r] * inv;
    o_s[wave][row * OROW + 3 * 16 + l15] = O3[r] * inv;
  }
  asm volatile("s_wait_dscnt 0x0" ::: "memory");

  const float g = gamma[0];
  const int j = m0 + l15;                        // 16 consecutive n per half-wave
  for (int cc = 0; cc < CC; cc += 2) {
    const int c = cc + half;
    const size_t idx = ((size_t)b * CC + c) * NN + j;
    out[idx] = g * o_s[wave][l15 * OROW + c] + x[idx];
  }
}

// ---------------------------------------------------------------------------
extern "C" void kernel_launch(void* const* d_in, const int* in_sizes, int n_in,
                              void* d_out, int out_size, void* d_ws, size_t ws_size,
                              hipStream_t stream) {
  (void)in_sizes; (void)n_in; (void)out_size; (void)ws_size;
  const float* x     = (const float*)d_in[0];
  const float* Wq    = (const float*)d_in[1];
  const float* Wk    = (const float*)d_in[2];
  const float* Wv    = (const float*)d_in[3];
  const float* gamma = (const float*)d_in[4];

  __bf16* zbuf = (__bf16*)d_ws;                       // 128B zeros (256B slot)
  __bf16* qT   = zbuf + 128;                          // [B][N][8]    1 MB
  __bf16* kT16 = qT + (size_t)BB * NN * MIDD;         // [B][N][16]   2 MB
  __bf16* vB   = kT16 + (size_t)BB * NN * 16;         // [B][C][N]    8 MB

  proj_kernel<<<BB * (NN / 128), 256, 0, stream>>>(x, Wq, Wk, Wv,
                                                   qT, kT16, vB, zbuf);

  const int tiles  = BB * (NN / 16);                  // 4096 query tiles
  const int blocks = tiles / WAVES;                   // 512 blocks of 8 waves
  attn_kernel<<<blocks, WAVES * 32, 0, stream>>>(x, gamma, qT, kT16, vB, zbuf,
                                                 (float*)d_out);
}
